// DecoderSelfAttention_23244363006596
// MI455X (gfx1250) — matrix operand order repaired and checked by reading, not compiled, not run on hardware
//
#include <hip/hip_runtime.h>
#include <hip/hip_bf16.h>

#define B_ 8
#define S_ 2048
#define H_ 1024

typedef __attribute__((ext_vector_type(16))) __bf16 v16bf;
typedef __attribute__((ext_vector_type(8)))  float  v8f;
typedef __attribute__((ext_vector_type(4)))  unsigned int u32x4;
typedef __attribute__((ext_vector_type(8)))  unsigned int u32x8;

union FragU { uint4 u[2]; v16bf v; };

// ---- CDNA5 async / tensor data movers ----

// 16B global -> LDS, tracked by ASYNCcnt, no VGPR round-trip.
__device__ __forceinline__ void async_copy16(void* lds, const void* g) {
  unsigned l = (unsigned)(size_t)lds;  // flat LDS aperture: addr[31:0] == LDS byte offset
  asm volatile("global_load_async_to_lds_b128 %0, %1, off" :: "v"(l), "v"(g) : "memory");
}
__device__ __forceinline__ void wait_async_le3() {
  asm volatile("s_wait_asynccnt 0x3" ::: "memory");
}
__device__ __forceinline__ void wait_async_0() {
  asm volatile("s_wait_asynccnt 0x0" ::: "memory");
}

// ---- WMMA fragment helpers ----

// A-fragment (16x32 bf16, MxK): lanes 0-15 rows M holding K 0..15,
// lanes 16-31 rows M holding K 16..31. Tile row-major [M][K].
__device__ __forceinline__ v16bf frag_a_lds(const __bf16* base, int pitch) {
  int lane = threadIdx.x & 31;
  int half = lane >> 4;
  int l = lane & 15;
  const __bf16* p = base + l * pitch + half * 16;
  FragU f;
  f.u[0] = *(const uint4*)(p);
  f.u[1] = *(const uint4*)(p + 8);
  return f.v;
}
// B-fragment (32x16 bf16, KxN): lanes 0-15 col N holding K 0..15, lanes 16-31 K 16..31.
// Tile row-major [N][K].
__device__ __forceinline__ v16bf frag_b_lds(const __bf16* base, int pitch) {
  int lane = threadIdx.x & 31;
  int half = lane >> 4;
  int l = lane & 15;
  const __bf16* p = base + l * pitch + half * 16;
  FragU f;
  f.u[0] = *(const uint4*)(p);
  f.u[1] = *(const uint4*)(p + 8);
  return f.v;
}
__device__ __forceinline__ v16bf frag_b_glb(const __bf16* base, long pitch) {
  int lane = threadIdx.x & 31;
  int half = lane >> 4;
  int l = lane & 15;
  const __bf16* p = base + (long)l * pitch + half * 16;
  FragU f;
  f.u[0] = *(const uint4*)(p);
  f.u[1] = *(const uint4*)(p + 8);
  return f.v;
}

__device__ __forceinline__ v8f wmma_bf16(v16bf a, v16bf b, v8f c) {
  return __builtin_amdgcn_wmma_f32_16x16x32_bf16(false, a, false, b, (short)0, c, false, false);
}

// ---------------- converts ----------------

__global__ void __launch_bounds__(256) cvt_bf16(const float* __restrict__ in,
                                                __bf16* __restrict__ out, int n) {
  for (int i = blockIdx.x * 256 + threadIdx.x; i < n; i += gridDim.x * 256)
    out[i] = (__bf16)in[i];
}

__global__ void __launch_bounds__(256) cvtT_bf16(const float* __restrict__ in,
                                                 __bf16* __restrict__ out) {
  int i = blockIdx.x * 256 + threadIdx.x;
  if (i < H_ * H_) {
    int n = i >> 10;
    int k = i & (H_ - 1);
    out[i] = (__bf16)in[k * H_ + n];
  }
}

// ------- shared 64x128 GEMM core: async double-buffered staging + WMMA -------
// A: MxK row-major, Bt: NxK row-major.

__device__ __forceinline__ void stage_async(const __bf16* __restrict__ A, int ldA,
                                            const __bf16* __restrict__ Bt, int ldB,
                                            int m0, int n0, int k0,
                                            __bf16* sA, __bf16* sB) {
  int tid = threadIdx.x;
  {  // A tile: 64 rows x 32 k  (256 x 16B chunks, 1 per thread)
    int row = tid >> 2, q = tid & 3;
    async_copy16(&sA[row * 40 + q * 8], A + (long)(m0 + row) * ldA + k0 + q * 8);
  }
  {  // B tile: 128 rows x 32 k (512 chunks, 2 per thread)
    int row = tid >> 2, q = tid & 3;
    async_copy16(&sB[row * 40 + q * 8], Bt + (long)(n0 + row) * ldB + k0 + q * 8);
    row += 64;
    async_copy16(&sB[row * 40 + q * 8], Bt + (long)(n0 + row) * ldB + k0 + q * 8);
  }
}

__device__ __forceinline__ void gemm_core(const __bf16* __restrict__ A, int ldA,
                                          const __bf16* __restrict__ Bt, int ldB,
                                          int m0, int n0, int ktiles,
                                          __bf16 (*sA)[64 * 40], __bf16 (*sB)[128 * 40],
                                          v8f acc[2][2]) {
  const int wid = threadIdx.x >> 5;
  const int wm = wid >> 2, wn = wid & 3;
  v8f z = {0.f, 0.f, 0.f, 0.f, 0.f, 0.f, 0.f, 0.f};
#pragma unroll
  for (int ms = 0; ms < 2; ++ms)
#pragma unroll
    for (int ns = 0; ns < 2; ++ns) acc[ms][ns] = z;

  stage_async(A, ldA, Bt, ldB, m0, n0, 0, sA[0], sB[0]);

  for (int kt = 0; kt < ktiles; ++kt) {
    int cur = kt & 1;
    if (kt + 1 < ktiles) {
      stage_async(A, ldA, Bt, ldB, m0, n0, (kt + 1) * 32, sA[cur ^ 1], sB[cur ^ 1]);
      wait_async_le3();  // current tile's 3 chunks retired (in-order completion)
    } else {
      wait_async_0();
    }
    __syncthreads();
    v16bf af[2], bf[2];
#pragma unroll
    for (int ms = 0; ms < 2; ++ms)
      af[ms] = frag_a_lds(sA[cur] + (wm * 32 + ms * 16) * 40, 40);
#pragma unroll
    for (int ns = 0; ns < 2; ++ns)
      bf[ns] = frag_b_lds(sB[cur] + (wn * 32 + ns * 16) * 40, 40);
#pragma unroll
    for (int ms = 0; ms < 2; ++ms)
#pragma unroll
      for (int ns = 0; ns < 2; ++ns)
        acc[ms][ns] = wmma_bf16(af[ms], bf[ns], acc[ms][ns]);
    __syncthreads();
  }
}

// ---------------- K1: QKV projection ----------------

__global__ void __launch_bounds__(256) qkv_gemm(const __bf16* __restrict__ X,
                                                const __bf16* __restrict__ Wt,
                                                const float* __restrict__ bias,
                                                __bf16* Crow, __bf16* Ctr, float* Cf) {
  __shared__ __bf16 sA[2][64 * 40];
  __shared__ __bf16 sB[2][128 * 40];
  int m0 = blockIdx.y * 64;
  int n0 = blockIdx.x * 128;
  v8f acc[2][2];
  gemm_core(X, H_, Wt, H_, m0, n0, H_ / 32, sA, sB, acc);

  int lane = threadIdx.x & 31;
  int wid = threadIdx.x >> 5;
  int wm = wid >> 2, wn = wid & 3;
  int l = lane & 15, half = lane >> 4;
#pragma unroll
  for (int ms = 0; ms < 2; ++ms)
#pragma unroll
    for (int ns = 0; ns < 2; ++ns) {
      int n = n0 + wn * 32 + ns * 16 + l;
      float bv = bias[n];
#pragma unroll
      for (int r = 0; r < 8; ++r) {
        int m = m0 + wm * 32 + ms * 16 + r + 8 * half;
        float val = acc[ms][ns][r] + bv;
        if (Crow) Crow[(long)m * H_ + n] = (__bf16)val;
        if (Ctr) {  // transposed per-batch copy: vt[b][h][j]
          int bb = m >> 11;
          int j = m & (S_ - 1);
          Ctr[(long)bb * H_ * S_ + (long)n * S_ + j] = (__bf16)val;
        }
        if (Cf) Cf[(long)m * H_ + n] = val;
      }
    }
}

// ---------------- K2: causal scores + softmax -> probs (bf16) ----------------
// Q block staged by the Tensor Data Mover as 3D tile [32 chunks][16 rows][32 h],
// with D# padding giving an 80B (40 bf16) row pitch for bank-friendly ds_load_b128.

__global__ void __launch_bounds__(256) attn_scores_softmax(const __bf16* __restrict__ Q,
                                                           const __bf16* __restrict__ K,
                                                           __bf16* __restrict__ P) {
  __shared__ __bf16 sQ[32 * 16 * 40];  // 40,960 B
  __shared__ float red[8][16];
  int bb = blockIdx.y;
  int i0 = blockIdx.x * 16;
  const __bf16* Qb = Q + (long)bb * S_ * H_ + (long)i0 * H_;
  const __bf16* Kb = K + (long)bb * S_ * H_;
  __bf16* Pb = P + (long)bb * S_ * S_;
  int tid = threadIdx.x, lane = tid & 31, wid = tid >> 5;
  int l = lane & 15, half = lane >> 4;

  if (wid == 0) {
    // Tensor DMA descriptor (D#): element (x,y,z) at Qb + 2*(x + y*1024 + z*32),
    // tile 32 x 16 x 32, data_size=2B, pad 4 DWORDs after every 16 DWORDs.
    unsigned lds_base = (unsigned)(size_t)(&sQ[0]);
    unsigned long long ga = (unsigned long long)(size_t)Qb;
    u32x4 g0 = { 1u,                                   // count=1, user mode
                 lds_base,                             // lds_addr
                 (unsigned)ga,                         // global_addr[31:0]
                 (unsigned)((ga >> 32) & 0x01FFFFFFu) | (2u << 30) };  // ga[56:32], type=2
    u32x8 g1 = { (1u << 16) | (1u << 20) | (3u << 22) | (3u << 25),
                 //  data_size=2B | pad_enable | pad_interval=16DW | pad_amount=4DW
                 (32u << 16),          // tensor_dim0 = 32  (bits 79:48)
                 (16u << 16),          // tensor_dim1 = 16  (bits 111:80)
                 (32u << 16),          // tile_dim0 = 32    (bits 127:112)
                 16u | (32u << 16),    // tile_dim1=16, tile_dim2=32
                 1024u,                // tensor_dim0_stride = 1024 (bits 207:160)
                 (32u << 16),          // tensor_dim1_stride = 32   (bits 255:208)
                 0u };
    u32x4 g2 = { 32u, 0u, 0u, 0u };    // tensor_dim2 = 32
    u32x4 g3 = { 0u, 0u, 0u, 0u };
    asm volatile("tensor_load_to_lds %0, %1, %2, %3"
                 :: "s"(g0), "s"(g1), "s"(g2), "s"(g3) : "memory");
    __builtin_amdgcn_s_wait_tensorcnt(0);
  }
  __syncthreads();

  int jtiles = (i0 >> 4) + 1;              // tiles touching the causal region
  int jtiles_pad = ((i0 & ~63) >> 4) + 4;  // region the PV kernel will read
  v8f sacc[16];

#pragma unroll
  for (int t = 0; t < 16; ++t) {
    int jt = wid + 8 * t;
    if (jt < jtiles) {
      v8f acc = {0.f, 0.f, 0.f, 0.f, 0.f, 0.f, 0.f, 0.f};
      const __bf16* Kt = Kb + (long)(jt * 16) * H_;
      for (int hc = 0; hc < 32; ++hc) {
        v16bf a = frag_a_lds(sQ + hc * (16 * 40), 40);
        v16bf b = frag_b_glb(Kt + hc * 32, H_);
        acc = wmma_bf16(a, b, acc);
      }
      int sj = jt * 16 + l;
#pragma unroll
      for (int r = 0; r < 8; ++r) {
        int si = i0 + r + 8 * half;
        if (sj > si) acc[r] = -1.0e10f;  // mask: scores*mask - NEG*(1-mask)
      }
      sacc[t] = acc;
    }
  }

  // row-wise max
  float mx[8];
#pragma unroll
  for (int r = 0; r < 8; ++r) mx[r] = -3.0e38f;
#pragma unroll
  for (int t = 0; t < 16; ++t) {
    int jt = wid + 8 * t;
    if (jt < jtiles) {
#pragma unroll
      for (int r = 0; r < 8; ++r) mx[r] = fmaxf(mx[r], sacc[t][r]);
    }
  }
#pragma unroll
  for (int off = 1; off <= 8; off <<= 1)
#pragma unroll
    for (int r = 0; r < 8; ++r) mx[r] = fmaxf(mx[r], __shfl_xor(mx[r], off, 32));
  if (l == 0) {
#pragma unroll
    for (int r = 0; r < 8; ++r) red[wid][half * 8 + r] = mx[r];
  }
  __syncthreads();
#pragma unroll
  for (int r = 0; r < 8; ++r) {
    float m = -3.0e38f;
#pragma unroll
    for (int w = 0; w < 8; ++w) m = fmaxf(m, red[w][half * 8 + r]);
    mx[r] = m;
  }
  __syncthreads();

  // exp + row-wise sum
  float sm[8];
#pragma unroll
  for (int r = 0; r < 8; ++r) sm[r] = 0.f;
#pragma unroll
  for (int t = 0; t < 16; ++t) {
    int jt = wid + 8 * t;
    if (jt < jtiles) {
#pragma unroll
      for (int r = 0; r < 8; ++r) {
        float e = __expf(sacc[t][r] - mx[r]);
        sacc[t][r] = e;
        sm[r] += e;
      }
    }
  }
#pragma unroll
  for (int off = 1; off <= 8; off <<= 1)
#pragma unroll
    for (int r = 0; r < 8; ++r) sm[r] += __shfl_xor(sm[r], off, 32);
  if (l == 0) {
#pragma unroll
    for (int r = 0; r < 8; ++r) red[wid][half * 8 + r] = sm[r];
  }
  __syncthreads();
  float inv[8];
#pragma unroll
  for (int r = 0; r < 8; ++r) {
    float s = 0.f;
#pragma unroll
    for (int w = 0; w < 8; ++w) s += red[w][half * 8 + r];
    inv[r] = 1.f / s;  // diagonal always unmasked -> s > 0
  }

  // normalize + store probs (zero-fill up to 64-row diagonal block boundary)
#pragma unroll
  for (int t = 0; t < 16; ++t) {
    int jt = wid + 8 * t;
    if (jt < jtiles_pad) {
      if (jt < jtiles) {
#pragma unroll
        for (int r = 0; r < 8; ++r) {
          int m = i0 + r + 8 * half;
          Pb[(long)m * S_ + jt * 16 + l] = (__bf16)(sacc[t][r] * inv[r]);
        }
      } else {
#pragma unroll
        for (int r = 0; r < 8; ++r) {
          int m = i0 + r + 8 * half;
          Pb[(long)m * S_ + jt * 16 + l] = (__bf16)0.f;
        }
      }
    }
  }
}

// ---------------- K3: attn = probs @ v  (+ residual v) ----------------

__global__ void __launch_bounds__(256) attn_pv(const __bf16* __restrict__ P,
                                               const __bf16* __restrict__ Vt,
                                               const float* __restrict__ Vf,
                                               float* __restrict__ Out) {
  __shared__ __bf16 sA[2][64 * 40];
  __shared__ __bf16 sB[2][128 * 40];
  int bb = blockIdx.z;
  int i0 = blockIdx.y * 64;
  int h0 = blockIdx.x * 128;
  const __bf16* A = P + (long)bb * S_ * S_;
  const __bf16* Bt = Vt + (long)bb * H_ * S_;
  v8f acc[2][2];
  gemm_core(A, S_, Bt, S_, i0, h0, 2 * blockIdx.y + 2, sA, sB, acc);

  int lane = threadIdx.x & 31;
  int wid = threadIdx.x >> 5;
  int wm = wid >> 2, wn = wid & 3;
  int l = lane & 15, half = lane >> 4;
#pragma unroll
  for (int ms = 0; ms < 2; ++ms)
#pragma unroll
    for (int ns = 0; ns < 2; ++ns) {
      int n = h0 + wn * 32 + ns * 16 + l;
#pragma unroll
      for (int r = 0; r < 8; ++r) {
        int m = i0 + wm * 32 + ms * 16 + r + 8 * half;
        long idx = ((long)bb * S_ + m) * H_ + n;
        Out[idx] = acc[ms][ns][r] + Vf[idx];
      }
    }
}

// ---------------- K4: in-place LayerNorm (one wave per row) ----------------

__global__ void __launch_bounds__(256) layernorm_inplace(float* __restrict__ Y,
                                                         const float* __restrict__ gamma,
                                                         const float* __restrict__ beta) {
  int wid = threadIdx.x >> 5, lane = threadIdx.x & 31;
  int row = blockIdx.x * 8 + wid;
  float* y = Y + (long)row * H_;
  float v[32];
  float s = 0.f, s2 = 0.f;
#pragma unroll
  for (int i = 0; i < 32; ++i) {
    float t = y[lane + 32 * i];
    v[i] = t;
    s += t;
    s2 += t * t;
  }
#pragma unroll
  for (int off = 1; off <= 16; off <<= 1) {
    s += __shfl_xor(s, off, 32);
    s2 += __shfl_xor(s2, off, 32);
  }
  float mu = s * (1.f / H_);
  float var = s2 * (1.f / H_) - mu * mu;
  float rs = rsqrtf(var + 1e-5f);
#pragma unroll
  for (int i = 0; i < 32; ++i) {
    int h = lane + 32 * i;
    y[h] = (v[i] - mu) * rs * gamma[h] + beta[h];
  }
}

// ---------------- launch ----------------

extern "C" void kernel_launch(void* const* d_in, const int* in_sizes, int n_in,
                              void* d_out, int out_size, void* d_ws, size_t ws_size,
                              hipStream_t stream) {
  (void)in_sizes; (void)n_in; (void)out_size; (void)ws_size;
  const float* x     = (const float*)d_in[0];
  const float* Wq    = (const float*)d_in[1];
  const float* bq    = (const float*)d_in[2];
  const float* Wk    = (const float*)d_in[3];
  const float* bk    = (const float*)d_in[4];
  const float* Wv    = (const float*)d_in[5];
  const float* bv    = (const float*)d_in[6];
  const float* gamma = (const float*)d_in[7];
  const float* beta  = (const float*)d_in[8];
  float* out = (float*)d_out;

  char* ws = (char*)d_ws;
  __bf16* xb  = (__bf16*)(ws + 0);           // 32 MB
  __bf16* qb  = (__bf16*)(ws + 33554432);    // 32 MB
  __bf16* kb  = (__bf16*)(ws + 67108864);    // 32 MB
  __bf16* vt  = (__bf16*)(ws + 100663296);   // 32 MB (per-batch transposed v)
  float*  vf  = (float*) (ws + 134217728);   // 64 MB
  __bf16* pr  = (__bf16*)(ws + 201326592);   // 64 MB (probs)
  __bf16* wqt = (__bf16*)(ws + 268435456);   // 2 MB
  __bf16* wkt = (__bf16*)(ws + 270532608);   // 2 MB
  __bf16* wvt = (__bf16*)(ws + 272629760);   // 2 MB

  cvt_bf16<<<8192, 256, 0, stream>>>(x, xb, B_ * S_ * H_);
  cvtT_bf16<<<4096, 256, 0, stream>>>(Wq, wqt);
  cvtT_bf16<<<4096, 256, 0, stream>>>(Wk, wkt);
  cvtT_bf16<<<4096, 256, 0, stream>>>(Wv, wvt);

  dim3 g1(H_ / 128, (B_ * S_) / 64);
  qkv_gemm<<<g1, 256, 0, stream>>>(xb, wqt, bq, qb, nullptr, nullptr);
  qkv_gemm<<<g1, 256, 0, stream>>>(xb, wkt, bk, kb, nullptr, nullptr);
  qkv_gemm<<<g1, 256, 0, stream>>>(xb, wvt, bv, nullptr, vt, vf);

  dim3 g2(S_ / 16, B_);
  attn_scores_softmax<<<g2, 256, 0, stream>>>(qb, kb, pr);

  dim3 g3(H_ / 128, S_ / 64, B_);
  attn_pv<<<g3, 256, 0, stream>>>(pr, vt, vf, out);

  layernorm_inplace<<<(B_ * S_) / 8, 256, 0, stream>>>(out, gamma, beta);
}
